// DifferntialAttention_91036126806450
// MI455X (gfx1250) — compile-verified, matrix operand order checked by reading, not run on hardware
//
#include <hip/hip_runtime.h>

// ---------------- problem constants (from reference) ----------------
constexpr int kB  = 2;
constexpr int kS  = 2048;
constexpr int kE  = 2048;
constexpr int kH  = 16;
constexpr int kHD = 128;            // kE / kH
constexpr float kScaler = 0.08838834764831845f;   // 128^-0.5
constexpr float kInitLambda = 0.8f;

// ---------------- WMMA types / helpers ----------------
typedef __attribute__((ext_vector_type(16))) __bf16 v16bf;
typedef __attribute__((ext_vector_type(8)))  float  v8f;

__device__ __forceinline__ unsigned short bfbits(float f) {
    unsigned u = __builtin_bit_cast(unsigned, f);
    unsigned r = u + 0x7FFFu + ((u >> 16) & 1u);       // RNE
    return (unsigned short)(r >> 16);
}
__device__ __forceinline__ __bf16 f2bf(float f) {
    unsigned short h = bfbits(f);
    return __builtin_bit_cast(__bf16, h);
}
__device__ __forceinline__ float bf2f(__bf16 b) {
    unsigned short h = __builtin_bit_cast(unsigned short, b);
    unsigned u = ((unsigned)h) << 16;
    return __builtin_bit_cast(float, u);
}
__device__ __forceinline__ unsigned pack2bf(float lo, float hi) {
    return (unsigned)bfbits(lo) | ((unsigned)bfbits(hi) << 16);
}

__device__ __forceinline__ void store_c(float* C, size_t idx, float v)  { C[idx] = v; }
__device__ __forceinline__ void store_c(__bf16* C, size_t idx, float v) { C[idx] = f2bf(v); }

// Load one 16x32 bf16 fragment (A-or-B layout; both read rows of a row-major
// [16][K] tile with the CDNA5 per-lane K pattern: lane L holds row L%16,
// K = (L/16)*8 + {0..7} in elems 0..7 and 16 + (L/16)*8 + {0..7} in elems 8..15).
__device__ __forceinline__ v16bf lds_load_frag(const __bf16* __restrict__ base, int stride) {
    const int lane = threadIdx.x & 31;
    const __bf16* p = base + (lane & 15) * stride + (lane >> 4) * 8;
    v16bf v;
#pragma unroll
    for (int i = 0; i < 8; ++i) { v[i] = p[i]; v[8 + i] = p[16 + i]; }
    return v;
}

__device__ __forceinline__ v8f wmma_bf16(v16bf a, v16bf b, v8f c) {
    return __builtin_amdgcn_wmma_f32_16x16x32_bf16(false, a, false, b, (short)0, c, false, false);
}

// ---------------- fp32 -> bf16 bulk conversion (8 elems / thread) ----------------
__global__ __launch_bounds__(256) void cvt_bf16_kernel(
    const float* __restrict__ src, __bf16* __restrict__ dst)
{
    size_t base = ((size_t)blockIdx.x * 256 + threadIdx.x) * 8;
    float4 a = *(const float4*)(src + base);
    float4 b = *(const float4*)(src + base + 4);
    uint4 w;
    w.x = pack2bf(a.x, a.y);
    w.y = pack2bf(a.z, a.w);
    w.z = pack2bf(b.x, b.y);
    w.w = pack2bf(b.z, b.w);
    *(uint4*)(dst + base) = w;
}

// ---------------- bf16 WMMA GEMM: C[M,N] = A[M,K] * B[N,K]^T ----------------
// M % 128 == 0, N % 128 == 0, K % 32 == 0.
// Register + LDS double-buffered staging: one block barrier per k-step.
template <typename CT>
__global__ __launch_bounds__(256) void gemm_bf16_wmma_kernel(
    const __bf16* __restrict__ A, const __bf16* __restrict__ Bm,
    CT* __restrict__ C, int M, int N, int K)
{
    constexpr int BK = 32, LDT = 40;                   // pad 40 elems -> 80B, 16B-aligned
    __shared__ __bf16 As[2][128 * LDT];                // 2 x 10 KB
    __shared__ __bf16 Bs[2][128 * LDT];                // 2 x 10 KB

    const int tid  = threadIdx.x;
    const int wave = tid >> 5;
    const int lane = tid & 31;
    const int m0 = blockIdx.y * 128;
    const int n0 = blockIdx.x * 128;
    const int tmB = (wave >> 1) * 2;                   // tile-row base: 0,2,4,6
    const int tnB = (wave & 1) * 4;                    // tile-col base: 0 or 4

    // staging: 512 16B-chunks per matrix, 2 per thread
    const int c0 = tid, c1 = tid + 256;
    const int r0 = c0 >> 2, kkA0 = (c0 & 3) * 8;
    const int r1 = c1 >> 2, kkA1 = (c1 & 3) * 8;
    const __bf16* Aptr = A + (size_t)m0 * K;
    const __bf16* Bptr = Bm + (size_t)n0 * K;

    // tile 0 -> LDS buffer 0
    {
        uint4 pa0 = *(const uint4*)(Aptr + (size_t)r0 * K + kkA0);
        uint4 pa1 = *(const uint4*)(Aptr + (size_t)r1 * K + kkA1);
        uint4 pb0 = *(const uint4*)(Bptr + (size_t)r0 * K + kkA0);
        uint4 pb1 = *(const uint4*)(Bptr + (size_t)r1 * K + kkA1);
        *(uint4*)&As[0][r0 * LDT + kkA0] = pa0;
        *(uint4*)&As[0][r1 * LDT + kkA1] = pa1;
        *(uint4*)&Bs[0][r0 * LDT + kkA0] = pb0;
        *(uint4*)&Bs[0][r1 * LDT + kkA1] = pb1;
    }
    __syncthreads();

    v8f acc[2][4];
#pragma unroll
    for (int i = 0; i < 2; ++i)
#pragma unroll
        for (int j = 0; j < 4; ++j)
#pragma unroll
            for (int e = 0; e < 8; ++e) acc[i][j][e] = 0.0f;

    int p = 0;
    for (int k0 = 0; k0 < K; k0 += BK) {
        const int kn = k0 + BK;
        uint4 pa0, pa1, pb0, pb1;
        if (kn < K) {                                  // global loads for next tile
            pa0 = *(const uint4*)(Aptr + (size_t)r0 * K + kn + kkA0);
            pa1 = *(const uint4*)(Aptr + (size_t)r1 * K + kn + kkA1);
            pb0 = *(const uint4*)(Bptr + (size_t)r0 * K + kn + kkA0);
            pb1 = *(const uint4*)(Bptr + (size_t)r1 * K + kn + kkA1);
            if (kn + BK < K) {                         // L2/L0 prefetch 2 tiles ahead
                __builtin_prefetch(Aptr + (size_t)r0 * K + kn + BK + kkA0, 0, 0);
                __builtin_prefetch(Bptr + (size_t)r0 * K + kn + BK + kkA0, 0, 0);
            }
        }

        v16bf af[2], bfv[4];
#pragma unroll
        for (int i = 0; i < 2; ++i) af[i]  = lds_load_frag(&As[p][(tmB + i) * 16 * LDT], LDT);
#pragma unroll
        for (int j = 0; j < 4; ++j) bfv[j] = lds_load_frag(&Bs[p][(tnB + j) * 16 * LDT], LDT);
#pragma unroll
        for (int i = 0; i < 2; ++i)
#pragma unroll
            for (int j = 0; j < 4; ++j) acc[i][j] = wmma_bf16(af[i], bfv[j], acc[i][j]);

        if (kn < K) {                                  // stage next tile into other buffer
            *(uint4*)&As[p ^ 1][r0 * LDT + kkA0] = pa0;
            *(uint4*)&As[p ^ 1][r1 * LDT + kkA1] = pa1;
            *(uint4*)&Bs[p ^ 1][r0 * LDT + kkA0] = pb0;
            *(uint4*)&Bs[p ^ 1][r1 * LDT + kkA1] = pb1;
            __syncthreads();                           // single barrier per k-step
            p ^= 1;
        }
    }

    const int cn = lane & 15;
    const int rh = (lane >> 4) * 8;
#pragma unroll
    for (int i = 0; i < 2; ++i)
#pragma unroll
        for (int j = 0; j < 4; ++j)
#pragma unroll
            for (int e = 0; e < 8; ++e) {
                int row = m0 + (tmB + i) * 16 + rh + e;
                int col = n0 + (tnB + j) * 16 + cn;
                store_c(C, (size_t)row * N + col, acc[i][j][e]);
            }
}

// ---------------- lambda scalar ----------------
__global__ __launch_bounds__(128) void lambda_kernel(
    const float* __restrict__ lq1, const float* __restrict__ lq2,
    const float* __restrict__ lk1, const float* __restrict__ lk2,
    float* __restrict__ lam)
{
    __shared__ float s1[128], s2[128];
    int t = threadIdx.x;
    s1[t] = lq1[t] * lk1[t];
    s2[t] = lq2[t] * lk2[t];
    __syncthreads();
    for (int k = 64; k > 0; k >>= 1) {
        if (t < k) { s1[t] += s1[t + k]; s2[t] += s2[t + k]; }
        __syncthreads();
    }
    if (t == 0) *lam = __expf(s1[0]) - __expf(s2[0]) + kInitLambda;
}

// ---------------- RoPE + head split: qkv[B,S,3E] bf16 -> q/k/v [B*H,S,128] bf16 ----------------
__global__ __launch_bounds__(256) void rope_split_kernel(
    const __bf16* __restrict__ qkv,
    __bf16* __restrict__ qb, __bf16* __restrict__ kb, __bf16* __restrict__ vb)
{
    int idx = blockIdx.x * 256 + threadIdx.x;   // B*H*S*64 total
    int d  = idx & 63;
    int s  = (idx >> 6) & (kS - 1);
    int bh = idx >> 17;                          // 6 + 11 bits
    int b  = bh >> 4, h = bh & 15;

    size_t rowIn = ((size_t)(b * kS + s)) * (3 * kE);
    size_t qi = rowIn + (size_t)h * kHD;
    size_t ki = rowIn + kE + (size_t)h * kHD;
    size_t vi = rowIn + 2 * kE + (size_t)h * kHD;

    // inv_freq = 10000^(-2d/128) = exp(-d * ln(10000)/64)
    float inv = __expf(-(float)d * (9.210340371976184f / 64.0f));
    float ang = (float)s * inv;
    float sn, cs;
    __sincosf(ang, &sn, &cs);

    float qlo = bf2f(qkv[qi + d]), qhi = bf2f(qkv[qi + d + 64]);
    float klo = bf2f(qkv[ki + d]), khi = bf2f(qkv[ki + d + 64]);

    size_t rowOut = ((size_t)bh * kS + s) * kHD;
    qb[rowOut + d]      = f2bf(qlo * cs - qhi * sn);
    qb[rowOut + d + 64] = f2bf(qhi * cs + qlo * sn);
    kb[rowOut + d]      = f2bf(klo * cs - khi * sn);
    kb[rowOut + d + 64] = f2bf(khi * cs + klo * sn);
    vb[rowOut + d]      = qkv[vi + d];
    vb[rowOut + d + 64] = qkv[vi + d + 64];
}

// ---------------- dual-stream flash attention ----------------
// grid: (S/64, B*H), block 256 (8 waves).  out: fp32 [B,S,E] flat.
__global__ __launch_bounds__(256) void diffattn_kernel(
    const __bf16* __restrict__ qb, const __bf16* __restrict__ kb,
    const __bf16* __restrict__ vb, const float* __restrict__ lamp,
    float* __restrict__ out)
{
    constexpr int Br = 64, Bc = 32;
    constexpr int LQ = 72, LK = 72, LV = 40, LS = 36, LP = 40;

    __shared__ __bf16 Qs[2][Br * LQ];     // two streams (first/second half of head dim)
    __shared__ __bf16 Ks[2][Bc * LK];
    __shared__ __bf16 Vt[kHD * LV];       // V^T: [n][k]
    __shared__ float  Sc[Br * LS];
    __shared__ __bf16 Ps[Br * LP];
    __shared__ float  mrow[2][Br], lrow[2][Br], frow[2][Br];

    const int tid  = threadIdx.x;
    const int wave = tid >> 5;
    const int lane = tid & 31;
    const int q0 = blockIdx.x * Br;
    const int bh = blockIdx.y;
    const int b  = bh >> 4, h = bh & 15;
    const size_t headBase = (size_t)bh * kS * kHD;

    // ---- stage Q (both streams), 16B chunks: 1024 chunks, 4/thread ----
#pragma unroll
    for (int t = 0; t < 4; ++t) {
        int c = tid + t * 256;
        int r = c >> 4, d8 = (c & 15) * 8;
        uint4 v = *(const uint4*)(qb + headBase + (size_t)(q0 + r) * kHD + d8);
        if (d8 < 64) *(uint4*)&Qs[0][r * LQ + d8]      = v;
        else         *(uint4*)&Qs[1][r * LQ + d8 - 64] = v;
    }
    if (tid < Br) {
        mrow[0][tid] = -1e30f; mrow[1][tid] = -1e30f;
        lrow[0][tid] = 0.0f;   lrow[1][tid] = 0.0f;
    }

    const int tq  = wave >> 1;          // QK tile row (0..3), also PV tile row
    const int tk  = wave & 1;           // QK tile col (0..1)
    const int tnP = (wave & 1) * 4;     // PV tile-col base (0 or 4)
    const int rh  = (lane >> 4) * 8;
    const int cn  = lane & 15;

    v8f O[2][4];
#pragma unroll
    for (int st = 0; st < 2; ++st)
#pragma unroll
        for (int j = 0; j < 4; ++j)
#pragma unroll
            for (int e = 0; e < 8; ++e) O[st][j][e] = 0.0f;

    __syncthreads();

    for (int kc = 0; kc < kS; kc += Bc) {
        // ---- stage K (both streams): 512 chunks, 2/thread ----
#pragma unroll
        for (int t = 0; t < 2; ++t) {
            int c = tid + t * 256;
            int r = c >> 4, d8 = (c & 15) * 8;
            uint4 v = *(const uint4*)(kb + headBase + (size_t)(kc + r) * kHD + d8);
            if (d8 < 64) *(uint4*)&Ks[0][r * LK + d8]      = v;
            else         *(uint4*)&Ks[1][r * LK + d8 - 64] = v;
        }
        // ---- stage V^T: 512 chunks, 2/thread; b128 load + 8 b16 scatter ----
#pragma unroll
        for (int t = 0; t < 2; ++t) {
            int c = tid + t * 256;
            int kk = c >> 4, n8 = (c & 15) * 8;
            union { uint4 v; __bf16 h[8]; } u;
            u.v = *(const uint4*)(vb + headBase + (size_t)(kc + kk) * kHD + n8);
#pragma unroll
            for (int i = 0; i < 8; ++i) Vt[(n8 + i) * LV + kk] = u.h[i];
        }
        __syncthreads();

        for (int st = 0; st < 2; ++st) {
            // ---- scores: S = Q_st @ K_st^T  (K-dim 64 -> two bf16 k-steps) ----
            {
                v8f sacc;
#pragma unroll
                for (int e = 0; e < 8; ++e) sacc[e] = 0.0f;
                v16bf a0 = lds_load_frag(&Qs[st][tq * 16 * LQ], LQ);
                v16bf a1 = lds_load_frag(&Qs[st][tq * 16 * LQ + 32], LQ);
                v16bf b0 = lds_load_frag(&Ks[st][tk * 16 * LK], LK);
                v16bf b1 = lds_load_frag(&Ks[st][tk * 16 * LK + 32], LK);
                sacc = wmma_bf16(a0, b0, sacc);
                sacc = wmma_bf16(a1, b1, sacc);
#pragma unroll
                for (int e = 0; e < 8; ++e) {
                    int row = tq * 16 + rh + e;
                    int col = tk * 16 + cn;
                    float sv = sacc[e] * kScaler +
                               (((kc + col) > (q0 + row)) ? 1.0f : 0.0f);  // triu(+1) bias
                    Sc[row * LS + col] = sv;
                }
            }
            __syncthreads();

            // ---- online softmax (4 lanes per row, wave32 shuffles) ----
            {
                int row = tid >> 2, sub = tid & 3;
                const float* sr = &Sc[row * LS + sub * 8];
                float mx = -1e30f;
#pragma unroll
                for (int j = 0; j < 8; ++j) mx = fmaxf(mx, sr[j]);
                mx = fmaxf(mx, __shfl_xor(mx, 1, 32));
                mx = fmaxf(mx, __shfl_xor(mx, 2, 32));
                float mo = mrow[st][row];
                float mn = fmaxf(mo, mx);
                float sum = 0.0f;
#pragma unroll
                for (int j = 0; j < 8; ++j) {
                    float p = __expf(sr[j] - mn);
                    Ps[row * LP + sub * 8 + j] = f2bf(p);
                    sum += p;
                }
                sum += __shfl_xor(sum, 1, 32);
                sum += __shfl_xor(sum, 2, 32);
                if (sub == 0) {
                    float f = __expf(mo - mn);
                    frow[st][row] = f;
                    lrow[st][row] = lrow[st][row] * f + sum;
                    mrow[st][row] = mn;
                }
            }
            __syncthreads();

            // ---- O_st = O_st*f + P @ V  (K-dim 32, one WMMA per 16x16 tile) ----
            {
                v16bf pa = lds_load_frag(&Ps[tq * 16 * LP], LP);
#pragma unroll
                for (int j = 0; j < 4; ++j) {
                    v16bf vf = lds_load_frag(&Vt[(tnP + j) * 16 * LV], LV);
                    v8f pv;
#pragma unroll
                    for (int e = 0; e < 8; ++e) pv[e] = 0.0f;
                    pv = wmma_bf16(pa, vf, pv);
#pragma unroll
                    for (int e = 0; e < 8; ++e) {
                        float f = frow[st][tq * 16 + rh + e];
                        O[st][j][e] = O[st][j][e] * f + pv[e];
                    }
                }
            }
        }
        __syncthreads();
    }

    // ---- epilogue: out = O1/l1 - lam*O2/l2, layout [b, s, h*128 + col] ----
    const float lam = *lamp;
#pragma unroll
    for (int j = 0; j < 4; ++j)
#pragma unroll
        for (int e = 0; e < 8; ++e) {
            int row = tq * 16 + rh + e;
            int col = (tnP + j) * 16 + cn;
            float o = O[0][j][e] / lrow[0][row] - lam * (O[1][j][e] / lrow[1][row]);
            out[((size_t)(b * kS + q0 + row)) * kE + (size_t)h * kHD + col] = o;
        }
}

// ---------------- groupnorm over the reference's reshaped layout ----------------
// Flat per-batch [S*E]; group = 2^18 consecutive elements (128 seq positions x E).
__global__ __launch_bounds__(256) void group_stats_kernel(
    const float* __restrict__ y, float* __restrict__ stats)
{
    __shared__ float rs[256], rq[256];
    const int g = blockIdx.x;                        // 0 .. B*16-1
    const float* p = y + (size_t)g * 262144;
    float s = 0.0f, q = 0.0f;
    for (int i = threadIdx.x * 4; i < 262144; i += 256 * 4) {
        float4 v = *(const float4*)(p + i);
        s += v.x + v.y + v.z + v.w;
        q += v.x * v.x + v.y * v.y + v.z * v.z + v.w * v.w;
    }
    rs[threadIdx.x] = s; rq[threadIdx.x] = q;
    __syncthreads();
    for (int k = 128; k > 0; k >>= 1) {
        if (threadIdx.x < k) { rs[threadIdx.x] += rs[threadIdx.x + k]; rq[threadIdx.x] += rq[threadIdx.x + k]; }
        __syncthreads();
    }
    if (threadIdx.x == 0) {
        float mean = rs[0] * (1.0f / 262144.0f);
        stats[g * 2]     = mean;
        stats[g * 2 + 1] = rq[0] * (1.0f / 262144.0f) - mean * mean;
    }
}

__global__ __launch_bounds__(256) void gnorm_apply_kernel(
    const float* __restrict__ y, const float* __restrict__ stats,
    const float* __restrict__ gamma, const float* __restrict__ beta,
    __bf16* __restrict__ yn)
{
    size_t t = ((size_t)blockIdx.x * 256 + threadIdx.x) * 4;   // 4 elems/thread
    int g = (int)(t >> 18);
    int i = (int)(t >> 11) & (kE - 1);               // gamma/beta index = flat // 2048 (same for quad)
    float mean = stats[g * 2], var = stats[g * 2 + 1];
    float rs = rsqrtf(var + 1e-5f);
    float gm = gamma[i] * rs * (1.0f - kInitLambda);
    float bt = (beta[i] - gamma[i] * mean * rs) * (1.0f - kInitLambda);
    float4 v = *(const float4*)(y + t);
    uint2 w;
    w.x = pack2bf(v.x * gm + bt, v.y * gm + bt);
    w.y = pack2bf(v.z * gm + bt, v.w * gm + bt);
    *(uint2*)(yn + t) = w;
}

// ---------------- launch ----------------
extern "C" void kernel_launch(void* const* d_in, const int* in_sizes, int n_in,
                              void* d_out, int out_size, void* d_ws, size_t ws_size,
                              hipStream_t stream) {
    (void)in_sizes; (void)n_in; (void)out_size; (void)ws_size;
    const float* x     = (const float*)d_in[0];
    const float* w_qkv = (const float*)d_in[1];
    const float* wo    = (const float*)d_in[2];
    const float* lq1   = (const float*)d_in[3];
    const float* lq2   = (const float*)d_in[4];
    const float* lk1   = (const float*)d_in[5];
    const float* lk2   = (const float*)d_in[6];
    const float* gamma = (const float*)d_in[7];
    const float* beta  = (const float*)d_in[8];

    char* w = (char*)d_ws;
    size_t off = 0;
    auto alloc = [&](size_t bytes) -> void* {
        void* p = w + off;
        off += (bytes + 255) & ~(size_t)255;
        return p;
    };
    const size_t BS  = (size_t)kB * kS;              // 4096
    const size_t BHS = (size_t)kB * kH * kS;         // 65536
    __bf16* xb    = (__bf16*)alloc(BS * kE * sizeof(__bf16));          // 17 MB
    __bf16* wqkvb = (__bf16*)alloc((size_t)3 * kE * kE * sizeof(__bf16)); // 25 MB
    __bf16* wob   = (__bf16*)alloc((size_t)kE * kE * sizeof(__bf16));  // 8 MB
    __bf16* qkvb  = (__bf16*)alloc(BS * 3 * kE * sizeof(__bf16));      // 50 MB
    __bf16* qb    = (__bf16*)alloc(BHS * kHD * sizeof(__bf16));        // 17 MB
    __bf16* kbuf  = (__bf16*)alloc(BHS * kHD * sizeof(__bf16));        // 17 MB
    __bf16* vbuf  = (__bf16*)alloc(BHS * kHD * sizeof(__bf16));        // 17 MB
    float*  attn  = (float*)alloc(BS * kE * sizeof(float));            // 34 MB
    __bf16* ynb   = (__bf16*)alloc(BS * kE * sizeof(__bf16));          // 17 MB
    float*  lam   = (float*)alloc(256);
    float*  stat  = (float*)alloc(256);

    // 0) one-time fp32 -> bf16 conversions (8 elems / thread)
    cvt_bf16_kernel<<<(int)(BS * kE / 2048), 256, 0, stream>>>(x, xb);
    cvt_bf16_kernel<<<(int)((size_t)3 * kE * kE / 2048), 256, 0, stream>>>(w_qkv, wqkvb);
    cvt_bf16_kernel<<<(int)((size_t)kE * kE / 2048), 256, 0, stream>>>(wo, wob);

    // 1) lambda scalar
    lambda_kernel<<<1, 128, 0, stream>>>(lq1, lq2, lk1, lk2, lam);

    // 2) qkv = x @ w_qkv^T   (bf16 in, bf16 out)  M=4096 N=6144 K=2048
    gemm_bf16_wmma_kernel<__bf16>
        <<<dim3(3 * kE / 128, BS / 128), 256, 0, stream>>>(xb, wqkvb, qkvb, (int)BS, 3 * kE, kE);

    // 3) RoPE + head split
    rope_split_kernel<<<(kB * kH * kS * 64) / 256, 256, 0, stream>>>(qkvb, qb, kbuf, vbuf);

    // 4) dual-stream flash attention
    diffattn_kernel<<<dim3(kS / 64, kB * kH), 256, 0, stream>>>(qb, kbuf, vbuf, lam, attn);

    // 5) group stats + normalize (+gamma/beta, x0.2) -> bf16
    group_stats_kernel<<<kB * 16, 256, 0, stream>>>(attn, stat);
    gnorm_apply_kernel<<<(int)((BS * kE) / 1024), 256, 0, stream>>>(attn, stat, gamma, beta, ynb);

    // 6) d_out = yn @ wo^T   (bf16 in, fp32 out)  M=4096 N=2048 K=2048
    gemm_bf16_wmma_kernel<float>
        <<<dim3(kE / 128, BS / 128), 256, 0, stream>>>(ynb, wob, (float*)d_out, (int)BS, kE, kE);
}
